// S4Model_6468220747779
// MI455X (gfx1250) — matrix operand order repaired and checked by reading, not compiled, not run on hardware
//
#include <hip/hip_runtime.h>
#include <hip/hip_bf16.h>
#include <math.h>

#define B_    4
#define L0_   8192
#define H_    768
#define H2_   1536
#define N_    32
#define NL_   6
#define NOUT_ 38

typedef __attribute__((ext_vector_type(8)))  __bf16 v8bf;
typedef __attribute__((ext_vector_type(16))) __bf16 v16bf;
typedef __attribute__((ext_vector_type(8)))  float  v8f;

// Build a 16-element A-fragment from two contiguous 8-element (16B) chunks:
// elements 0..7 = K(kb1+0..7), elements 8..15 = K(kb2+0..7)  [call sites: kb2 = kb1+8, contiguous k = 16*half+e]
__device__ __forceinline__ v16bf make_frag(const __bf16* p0, const __bf16* p1) {
  v8bf a = *(const v8bf*)p0;
  v8bf b = *(const v8bf*)p1;
  v16bf r;
#pragma unroll
  for (int e = 0; e < 8; ++e) { r[e] = a[e]; r[e + 8] = b[e]; }
  return r;
}

// ---------------- weight fp32 -> bf16 conversion ----------------
__global__ void k_cvt_w(const float* __restrict__ glu_w,
                        const float* __restrict__ enc_w2,
                        __bf16* __restrict__ wglu, __bf16* __restrict__ w2) {
  size_t i = (size_t)blockIdx.x * 256 + threadIdx.x;
  size_t ng = (size_t)NL_ * H2_ * H_;
  if (i < ng) wglu[i] = (__bf16)glu_w[i];
  if (i < (size_t)H_ * H_) w2[i] = (__bf16)enc_w2[i];
}

// ---------------- encoder layer 1 (8->H) + softsign, store bf16 (B,L,H) ----
__global__ void k_enc1(const float* __restrict__ xr, const float* __restrict__ w1,
                       const float* __restrict__ b1, __bf16* __restrict__ sbf) {
  int bl = blockIdx.x;                       // b*L0 + l
  int h  = blockIdx.y * 256 + threadIdx.x;   // 0..767
  float in[8];
#pragma unroll
  for (int k = 0; k < 8; ++k) in[k] = xr[(size_t)bl * 8 + k];
  const float* w = w1 + (size_t)h * 8;
  float acc = b1[h];
#pragma unroll
  for (int k = 0; k < 8; ++k) acc = fmaf(in[k], w[k], acc);
  acc = acc / (1.0f + fabsf(acc));           // softsign
  sbf[(size_t)bl * H_ + h] = (__bf16)acc;
}

// ---------------- encoder layer 2: x0(B,H,L) = W2(HxH) @ S^T, WMMA bf16 ----
__global__ void k_enc2(const __bf16* __restrict__ w2, const __bf16* __restrict__ sbf,
                       const float* __restrict__ b2, float* __restrict__ x0) {
  int lane = threadIdx.x;
  int n0 = blockIdx.x * 16, m0 = blockIdx.y * 32, b = blockIdx.z;
  int nn = lane & 15, half = lane >> 4;
  int col = n0 + nn;
  v8f c0 = {}, c1 = {};
  const __bf16* srow = sbf + ((size_t)b * L0_ + col) * H_;   // B[k,n]=s[col,k] contiguous in k
  for (int k0 = 0; k0 < H_; k0 += 32) {
    v16bf bf = *(const v16bf*)(srow + k0 + half * 16);        // K = k0+16*half+e  [contiguous order, same as A]
    int kb1 = k0 + half * 16, kb2 = k0 + half * 16 + 8;
    int rA0 = m0 + nn, rA1 = m0 + 16 + nn;
    v16bf a0 = make_frag(w2 + (size_t)rA0 * H_ + kb1, w2 + (size_t)rA0 * H_ + kb2);
    v16bf a1 = make_frag(w2 + (size_t)rA1 * H_ + kb1, w2 + (size_t)rA1 * H_ + kb2);
    c0 = __builtin_amdgcn_wmma_f32_16x16x32_bf16(false, a0, false, bf, (short)0, c0, false, false);
    c1 = __builtin_amdgcn_wmma_f32_16x16x32_bf16(false, a1, false, bf, (short)0, c1, false, false);
  }
#pragma unroll
  for (int r = 0; r < 8; ++r) {
    int row0 = m0 + r + 8 * half;
    int row1 = row0 + 16;
    x0[((size_t)b * H_ + row0) * L0_ + col] = c0[r] + b2[row0];
    x0[((size_t)b * H_ + row1) * L0_ + col] = c1[r] + b2[row1];
  }
}

// ---------------- bidirectional diagonal-SSM scan (replaces FFT conv) ------
// 2 waves per (b,h) row: wave0 forward, wave1 backward; lane = state n.
// 32 timesteps are batched per lane, then a 32x32 transpose-reduce (31
// shuffle+adds) leaves step lb+lane's cross-state sum in lane 'lane';
// a single conflict-free ds_add per lane accumulates into acc_s.
__global__ void k_scan(const float* __restrict__ x_in, const float* __restrict__ log_dt,
                       const float* __restrict__ A_log, const float* __restrict__ A_imp,
                       const float* __restrict__ C_rep, const float* __restrict__ C_imp,
                       const float* __restrict__ Dp, __bf16* __restrict__ ybf,
                       int layer, int L) {
  extern __shared__ float sm[];
  float* u_s   = sm;
  float* acc_s = sm + L;
  int bh = blockIdx.x;          // b*H + h
  int h  = bh % H_;
  const float* urow = x_in + (size_t)bh * L;
  int tid = threadIdx.x;
  for (int l = tid; l < L; l += 64) { u_s[l] = urow[l]; acc_s[l] = 0.0f; }
  __syncthreads();

  int wave = tid >> 5;          // 0 = forward(C[...,0]), 1 = backward(C[...,1])
  int n    = tid & 31;
  // per-(h,n) discretization: a = exp(dt*A), Cd = C*(exp(dt*A)-1)/A
  float dt  = expf(log_dt[layer * H_ + h]);
  size_t pidx = ((size_t)layer * H_ + h) * N_ + n;
  float Are = -expf(A_log[pidx]);
  float Aim = A_imp[pidx];
  float dre = dt * Are, dim = dt * Aim;
  float em  = expf(dre);
  float are = em * cosf(dim);
  float aim = em * sinf(dim);
  float e1re = are - 1.0f, e1im = aim;
  float den  = Are * Are + Aim * Aim;
  float fre  = (e1re * Are + e1im * Aim) / den;
  float fim  = (e1im * Are - e1re * Aim) / den;
  size_t cidx = (((size_t)layer * 2 + wave) * H_ + h) * N_ + n;
  float Crev = C_rep[cidx], Cimv = C_imp[cidx];
  float cd2re  =  2.0f * (Crev * fre - Cimv * fim);   // fold the 2x in
  float ncd2im = -2.0f * (Crev * fim + Cimv * fre);

  float sre = 0.0f, sim = 0.0f;
  if (wave == 0) {
    for (int lb = 0; lb < L; lb += 32) {
      float c[32];
#pragma unroll
      for (int j = 0; j < 32; ++j) {
        float u   = u_s[lb + j];
        float nre = fmaf(are, sre, fmaf(-aim, sim, u));
        float nim = fmaf(are, sim, aim * sre);
        sre = nre; sim = nim;
        c[j] = fmaf(cd2re, sre, ncd2im * sim);
      }
#pragma unroll
      for (int o = 16; o >= 1; o >>= 1) {
        bool hi = (n & o) != 0;
#pragma unroll
        for (int j = 0; j < o; ++j) {
          float mine = hi ? c[j + o] : c[j];
          float send = hi ? c[j] : c[j + o];
          c[j] = mine + __shfl_xor(send, o, 32);
        }
      }
      atomicAdd(&acc_s[lb + n], c[0]);   // lane n owns step lb+n
    }
  } else {
    for (int lb = L - 32; lb >= 0; lb -= 32) {
      float c[32];
#pragma unroll
      for (int j = 31; j >= 0; --j) {
        c[j] = fmaf(cd2re, sre, ncd2im * sim);      // state BEFORE u[l] (lag >= 1)
        float u   = u_s[lb + j];
        float nre = fmaf(are, sre, fmaf(-aim, sim, u));
        float nim = fmaf(are, sim, aim * sre);
        sre = nre; sim = nim;
      }
#pragma unroll
      for (int o = 16; o >= 1; o >>= 1) {
        bool hi = (n & o) != 0;
#pragma unroll
        for (int j = 0; j < o; ++j) {
          float mine = hi ? c[j + o] : c[j];
          float send = hi ? c[j] : c[j + o];
          c[j] = mine + __shfl_xor(send, o, 32);
        }
      }
      atomicAdd(&acc_s[lb + n], c[0]);
    }
  }
  __syncthreads();

  float Dv = Dp[layer * H_ + h];
  for (int l = tid; l < L; l += 64) {
    float u = u_s[l];
    float y = acc_s[l] + Dv * u;
    float g = 0.5f * y * (1.0f + erff(y * 0.70710678118654752f));   // exact GELU
    ybf[(size_t)bh * L + l] = (__bf16)g;
  }
}

// ---------------- LDS transpose: ybf (B,H,L) -> ytr (B,L,H), bf16 ----------
// 64x64 tiles so the GLU GEMM B-operand becomes contiguous-in-K (2 b128 loads
// per fragment instead of 16 scalar gathers).
__global__ void k_tr(const __bf16* __restrict__ ybf, __bf16* __restrict__ ytr, int L) {
  __shared__ __bf16 tile[64][65];
  int tid = threadIdx.x;
  int l0 = blockIdx.x * 64, h0 = blockIdx.y * 64, b = blockIdx.z;
  int row = tid >> 2;            // 0..63
  int c0  = (tid & 3) * 16;      // 0,16,32,48
  v16bf v = *(const v16bf*)(ybf + ((size_t)b * H_ + h0 + row) * L + l0 + c0);
#pragma unroll
  for (int e = 0; e < 16; ++e) tile[row][c0 + e] = v[e];
  __syncthreads();
  v16bf w;
#pragma unroll
  for (int e = 0; e < 16; ++e) w[e] = tile[c0 + e][row];
  *(v16bf*)(ytr + ((size_t)b * L + l0 + row) * H_ + h0 + c0) = w;
}

// ---------------- GLU GEMM (WMMA) + sigmoid gate + residual, fused --------
// Wave computes rows {m0..m0+32} and {H+m0..H+m0+32} (the gate pair) x 16 cols.
__global__ void k_glu(const __bf16* __restrict__ wglu, const __bf16* __restrict__ ytr,
                      const float* __restrict__ glu_b, const float* __restrict__ x_in,
                      float* __restrict__ g, int layer, int L) {
  int lane = threadIdx.x;
  int n0 = blockIdx.x * 16, m0 = blockIdx.y * 32, b = blockIdx.z;
  int nn = lane & 15, half = lane >> 4;
  int col = n0 + nn;
  const __bf16* wbase = wglu + (size_t)layer * H2_ * H_;
  const __bf16* yrow  = ytr + ((size_t)b * L + col) * H_;   // B[k,n]=Y^T[col,k], contiguous in k
  v8f cL0 = {}, cL1 = {}, cH0 = {}, cH1 = {};
  for (int k0 = 0; k0 < H_; k0 += 32) {
    v16bf bf = *(const v16bf*)(yrow + k0 + half * 16);
    int kb1 = k0 + half * 16, kb2 = k0 + half * 16 + 8;
    int r0 = m0 + nn, r1 = m0 + 16 + nn, r2 = H_ + m0 + nn, r3 = H_ + m0 + 16 + nn;
    v16bf a0 = make_frag(wbase + (size_t)r0 * H_ + kb1, wbase + (size_t)r0 * H_ + kb2);
    v16bf a1 = make_frag(wbase + (size_t)r1 * H_ + kb1, wbase + (size_t)r1 * H_ + kb2);
    v16bf a2 = make_frag(wbase + (size_t)r2 * H_ + kb1, wbase + (size_t)r2 * H_ + kb2);
    v16bf a3 = make_frag(wbase + (size_t)r3 * H_ + kb1, wbase + (size_t)r3 * H_ + kb2);
    cL0 = __builtin_amdgcn_wmma_f32_16x16x32_bf16(false, a0, false, bf, (short)0, cL0, false, false);
    cL1 = __builtin_amdgcn_wmma_f32_16x16x32_bf16(false, a1, false, bf, (short)0, cL1, false, false);
    cH0 = __builtin_amdgcn_wmma_f32_16x16x32_bf16(false, a2, false, bf, (short)0, cH0, false, false);
    cH1 = __builtin_amdgcn_wmma_f32_16x16x32_bf16(false, a3, false, bf, (short)0, cH1, false, false);
  }
  const float* gb = glu_b + (size_t)layer * H2_;
#pragma unroll
  for (int r = 0; r < 8; ++r) {
    int d0 = m0 + r + 8 * half;
    int d1 = d0 + 16;
    float vL0 = cL0[r] + gb[d0];
    float vH0 = cH0[r] + gb[H_ + d0];
    float vL1 = cL1[r] + gb[d1];
    float vH1 = cH1[r] + gb[H_ + d1];
    size_t o0 = ((size_t)b * H_ + d0) * L + col;
    size_t o1 = ((size_t)b * H_ + d1) * L + col;
    g[o0] = vL0 * (1.0f / (1.0f + expf(-vH0))) + x_in[o0];   // gate + residual
    g[o1] = vL1 * (1.0f / (1.0f + expf(-vH1))) + x_in[o1];
  }
}

// ---------------- LayerNorm over H (in (B,H,L) layout) + downsample -------
__global__ void k_ln(const float* __restrict__ g, const float* __restrict__ lnw,
                     const float* __restrict__ lnb, float* __restrict__ x_out,
                     int layer, int L, int Lout, int stride) {
  __shared__ float gbuf[16 * 769];
  __shared__ float ps[16][17];
  __shared__ float pq[16][17];
  __shared__ float mstat[16], rstat[16];
  int tid = threadIdx.x;
  int l = tid & 15, hc = tid >> 4;      // 16 l-positions x 16 h-chunks of 48
  int b = blockIdx.z;
  int lo = blockIdx.x * 16 + l;
  int li = lo * stride;
  float s = 0.0f, sq = 0.0f;
  for (int j = 0; j < 48; ++j) {
    int h = hc * 48 + j;
    float v = g[((size_t)b * H_ + h) * L + li];
    gbuf[l * 769 + h] = v;
    s += v; sq += v * v;
  }
  ps[l][hc] = s; pq[l][hc] = sq;
  __syncthreads();
  if (tid < 16) {
    float ts = 0.0f, tq = 0.0f;
    for (int c = 0; c < 16; ++c) { ts += ps[tid][c]; tq += pq[tid][c]; }
    float m = ts * (1.0f / H_);
    float var = tq * (1.0f / H_) - m * m;
    mstat[tid] = m; rstat[tid] = rsqrtf(var + 1e-5f);
  }
  __syncthreads();
  float m = mstat[l], rs = rstat[l];
  for (int j = 0; j < 48; ++j) {
    int h = hc * 48 + j;
    float v = (gbuf[l * 769 + h] - m) * rs * lnw[layer * H_ + h] + lnb[layer * H_ + h];
    x_out[((size_t)b * H_ + h) * Lout + lo] = v;
  }
}

// ---------------- output projection (H -> 38), tiny ------------------------
__global__ void k_out(const float* __restrict__ x, const float* __restrict__ ow,
                      const float* __restrict__ ob, float* __restrict__ out, int Lf) {
  int tid = threadIdx.x;
  int l  = blockIdx.x * 32 + (tid & 31);
  int og = tid >> 5;
  int b  = blockIdx.y;
  float acc[5] = {0.f, 0.f, 0.f, 0.f, 0.f};
  const float* xb = x + (size_t)b * H_ * Lf + l;
  for (int h = 0; h < H_; ++h) {
    float xv = xb[(size_t)h * Lf];
#pragma unroll
    for (int q = 0; q < 5; ++q) {
      int o = og * 5 + q;
      if (o < NOUT_) acc[q] = fmaf(xv, ow[(size_t)o * H_ + h], acc[q]);
    }
  }
#pragma unroll
  for (int q = 0; q < 5; ++q) {
    int o = og * 5 + q;
    if (o < NOUT_) out[((size_t)b * Lf + l) * NOUT_ + o] = acc[q] + ob[o];
  }
}

extern "C" void kernel_launch(void* const* d_in, const int* in_sizes, int n_in,
                              void* d_out, int out_size, void* d_ws, size_t ws_size,
                              hipStream_t stream) {
  (void)in_sizes; (void)n_in; (void)out_size; (void)ws_size;
  const float* x_raw  = (const float*)d_in[0];
  const float* enc_w1 = (const float*)d_in[1];
  const float* enc_b1 = (const float*)d_in[2];
  const float* enc_w2 = (const float*)d_in[3];
  const float* enc_b2 = (const float*)d_in[4];
  const float* log_dt = (const float*)d_in[5];
  const float* A_log  = (const float*)d_in[6];
  const float* A_imp  = (const float*)d_in[7];
  const float* C_rep  = (const float*)d_in[8];
  const float* C_imp  = (const float*)d_in[9];
  const float* Dp     = (const float*)d_in[10];
  const float* glu_w  = (const float*)d_in[11];
  const float* glu_b  = (const float*)d_in[12];
  const float* ln_w   = (const float*)d_in[13];
  const float* ln_b   = (const float*)d_in[14];
  const float* out_w  = (const float*)d_in[15];
  const float* out_b  = (const float*)d_in[16];

  char* ws = (char*)d_ws;
  size_t off = 0;
  __bf16* wglu = (__bf16*)(ws + off); off += (size_t)NL_ * H2_ * H_ * 2;
  __bf16* w2bf = (__bf16*)(ws + off); off += (size_t)H_ * H_ * 2;
  off = (off + 255) & ~(size_t)255;
  float*  bufA = (float*)(ws + off);  off += (size_t)B_ * H_ * L0_ * 4;        // 100.7 MB
  float*  bufB = (float*)(ws + off);  off += (size_t)B_ * H_ * (L0_ / 2) * 4;  // 50.3 MB
  __bf16* ybf  = (__bf16*)(ws + off); off += (size_t)B_ * H_ * L0_ * 2;        // 50.3 MB (also sbf)
  __bf16* ytr  = (__bf16*)(ws + off); off += (size_t)B_ * H_ * L0_ * 2;        // 50.3 MB transposed
  float*  gws  = (float*)(ws + off);  off += (size_t)B_ * H_ * L0_ * 4;        // 100.7 MB
  __bf16* sbf  = ybf;   // encoder hidden shares region with scan output

  {
    size_t ng = (size_t)NL_ * H2_ * H_;
    int blocks = (int)((ng + 255) / 256);
    k_cvt_w<<<blocks, 256, 0, stream>>>(glu_w, enc_w2, wglu, w2bf);
  }
  k_enc1<<<dim3(B_ * L0_, H_ / 256), 256, 0, stream>>>(x_raw, enc_w1, enc_b1, sbf);
  k_enc2<<<dim3(L0_ / 16, H_ / 32, B_), 32, 0, stream>>>(w2bf, sbf, enc_b2, bufA);

  float* xin  = bufA;
  float* xout = bufB;
  int L = L0_;
  for (int i = 0; i < NL_; ++i) {
    int stride = (i < 3) ? 2 : 1;
    int Lout = L / stride;
    k_scan<<<dim3(B_ * H_), 64, (size_t)2 * L * sizeof(float), stream>>>(
        xin, log_dt, A_log, A_imp, C_rep, C_imp, Dp, ybf, i, L);
    k_tr<<<dim3(L / 64, H_ / 64, B_), 256, 0, stream>>>(ybf, ytr, L);
    k_glu<<<dim3(L / 16, H_ / 32, B_), 32, 0, stream>>>(wglu, ytr, glu_b, xin, gws, i, L);
    k_ln<<<dim3(Lout / 16, 1, B_), 256, 0, stream>>>(gws, ln_w, ln_b, xout, i, L, Lout, stride);
    float* t = xin; xin = xout; xout = t;
    L = Lout;
  }
  k_out<<<dim3(L / 32, B_), 256, 0, stream>>>(xin, out_w, out_b, (float*)d_out, L);
}
